// TransformerBlock_27565100105936
// MI455X (gfx1250) — matrix operand order repaired and checked by reading, not compiled, not run on hardware
//
#include <hip/hip_runtime.h>
#include <hip/hip_bf16.h>
#include <math.h>

#define DIMD   512
#define SEQ    4096
#define BATCH  2
#define HEADS  8
#define HDIM   64
#define MLPD   2048
#define TOK    (BATCH * SEQ)   // 8192 rows in all GEMMs

typedef unsigned short u16b;
typedef __attribute__((ext_vector_type(16))) __bf16 v16bf;
typedef __attribute__((ext_vector_type(8)))  float  v8f;
typedef __attribute__((ext_vector_type(4)))  unsigned int uint32x4;
typedef __attribute__((ext_vector_type(8)))  int          int32x8;
typedef __attribute__((ext_vector_type(4)))  int          int32x4;

#if __has_builtin(__builtin_amdgcn_tensor_load_to_lds) && \
    __has_builtin(__builtin_amdgcn_s_wait_tensorcnt)
#define USE_TDM 1
#else
#define USE_TDM 0
#endif

union FragU { v16bf v; uint4 q[2]; };

__device__ inline v8f zero8() {
  v8f z = {0.f, 0.f, 0.f, 0.f, 0.f, 0.f, 0.f, 0.f};
  return z;
}

__device__ inline u16b f32_to_bf16(float f) {
  union { float f; unsigned u; } c; c.f = f;
  unsigned u = c.u;
  u += 0x7FFFu + ((u >> 16) & 1u);       // round-to-nearest-even
  return (u16b)(u >> 16);
}

// A fragment (16x32 bf16): row = lane&15; K bytes: VGPR0-3 = K(0..7 | 8..15), VGPR4-7 = +16
template <typename P>
__device__ inline v16bf load_a_frag(const P* base, int stride, int lane) {
  FragU f;
  const P* p = base + (lane & 15) * stride + ((lane & 16) ? 8 : 0);
  f.q[0] = *(const uint4*)(p);
  f.q[1] = *(const uint4*)(p + 16);
  return f.v;
}

// B fragment (32x16 bf16) from Bt[N][K]: col = lane&15; K: VGPR0-3 = K(0..7 | 8..15), VGPR4-7 = +16
template <typename P>
__device__ inline v16bf load_b_frag(const P* base, int stride, int lane) {
  FragU f;
  const P* p = base + (lane & 15) * stride + ((lane & 16) ? 8 : 0);
  f.q[0] = *(const uint4*)(p);
  f.q[1] = *(const uint4*)(p + 16);
  return f.v;
}

__device__ inline v8f wmma_bf16(v16bf a, v16bf b, v8f c) {
  return __builtin_amdgcn_wmma_f32_16x16x32_bf16(false, a, false, b, (short)0, c,
                                                 false, false);
}

#if USE_TDM
// LDS byte offset of a __shared__ pointer (ptrtoint of addrspace(3) pointer)
__device__ inline unsigned lds_addr_of(const void* p) {
  return (unsigned)(unsigned long long)(__attribute__((address_space(3))) const void*)p;
}

// 2D tile load via Tensor Data Mover (6-arg builtin on this toolchain).
//  - bf16 elements (data_size=1 -> 2 bytes)
//  - tile: 32 elems wide x tile_rows, row stride in memory = K_elems
//  - LDS padding: after each 64B row (16 DWORDs, code 3), add 16B (4 DWORDs, code 3)
//    => LDS row pitch = 40 bf16 elements, matching the fragment loaders.
__device__ inline void tdm_load_2d(unsigned lds_byte_addr, const void* gaddr,
                                   unsigned K_elems, unsigned tile_rows) {
  unsigned long long ga = (unsigned long long)gaddr;
  uint32x4 g0;
  g0[0] = 1u;                                              // count=1, user mode
  g0[1] = lds_byte_addr;                                   // lds_addr
  g0[2] = (unsigned)ga;                                    // global_addr[31:0]
  g0[3] = ((unsigned)(ga >> 32) & 0x01FFFFFFu) | (2u << 30); // addr[56:32] | type=2
  int32x8 g1;
  g1[0] = (int)((1u << 16)        // data_size = 2 bytes
              | (1u << 20)        // pad_enable
              | (3u << 22)        // pad_interval: 16 DWORDs (64B = one tile row)
              | (3u << 25));      // pad_amount: 4 DWORDs (16B = 8 elems)
  g1[1] = (int)((K_elems & 0xFFFFu) << 16);                // tensor_dim0[15:0]
  g1[2] = (int)((K_elems >> 16) | (8192u << 16));          // dim0[31:16] | dim1[15:0]
  g1[3] = (int)(32u << 16);                                // dim1[31:16]=0 | tile_dim0=32
  g1[4] = (int)tile_rows;                                  // tile_dim1 | tile_dim2=0
  g1[5] = (int)K_elems;                                    // tensor_dim0_stride[31:0]
  g1[6] = 0;                                               // stride0[47:32] | stride1 lo
  g1[7] = 0;
  int32x4 z4 = {0, 0, 0, 0};
  int32x8 z8 = {0, 0, 0, 0, 0, 0, 0, 0};
  __builtin_amdgcn_tensor_load_to_lds(g0, g1, z4, z4, z8, 0);
}
#endif

// ---------------- weight convert + transpose: dst[N][K] (bf16) = src[K][N] (f32)
__global__ __launch_bounds__(256) void cvt_t_kernel(const float* __restrict__ src,
                                                    u16b* __restrict__ dst,
                                                    int K, int N) {
  int idx = blockIdx.x * 256 + threadIdx.x;
  int kk = idx / N;
  int nn = idx - kk * N;
  dst[(size_t)nn * K + kk] = f32_to_bf16(src[idx]);
}

// ---------------- LayerNorm: one wave per 512-wide row, bf16 output
__global__ __launch_bounds__(256) void ln_kernel(const float* __restrict__ x,
                                                 const float* __restrict__ g,
                                                 const float* __restrict__ b,
                                                 u16b* __restrict__ out) {
  const int lane = threadIdx.x & 31;
  const int wave = threadIdx.x >> 5;
  const int row  = blockIdx.x * 8 + wave;
  const float* xr = x + (size_t)row * DIMD;
  float4 xv[4];
  float sum = 0.f, sq = 0.f;
#pragma unroll
  for (int i = 0; i < 4; ++i) {
    xv[i] = *(const float4*)&xr[lane * 16 + i * 4];
    sum += xv[i].x + xv[i].y + xv[i].z + xv[i].w;
    sq  += xv[i].x * xv[i].x + xv[i].y * xv[i].y + xv[i].z * xv[i].z + xv[i].w * xv[i].w;
  }
#pragma unroll
  for (int off = 1; off < 32; off <<= 1) {
    sum += __shfl_xor(sum, off);
    sq  += __shfl_xor(sq, off);
  }
  float mean = sum * (1.0f / DIMD);
  float var  = sq  * (1.0f / DIMD) - mean * mean;
  float rstd = rsqrtf(var + 1e-5f);
  u16b* orow = out + (size_t)row * DIMD;
#pragma unroll
  for (int i = 0; i < 4; ++i) {
    float vv[4] = {xv[i].x, xv[i].y, xv[i].z, xv[i].w};
#pragma unroll
    for (int j = 0; j < 4; ++j) {
      int col = lane * 16 + i * 4 + j;
      orow[col] = f32_to_bf16((vv[j] - mean) * rstd * g[col] + b[col]);
    }
  }
}

// ---------------- Tiled bf16 GEMM: C[8192,N] = A[8192,K] @ Bt[N,K]^T
// Block 256 thr = 8 waves (2x4), block tile 128x256, wave tile 64x64, K-chunk 32.
// Tiles staged by the Tensor Data Mover (double-buffered) when available.
// EPI: 0=Q scatter(*0.125), 1=K scatter, 2=V transposed scatter,
//      3=residual+bias (f32 out), 4=GELU(acc+bias) (bf16 out)
template <int EPI>
__global__ __launch_bounds__(256) void gemm_kernel(const u16b* __restrict__ A,
                                                   const u16b* __restrict__ Bt,
                                                   int K, int N,
                                                   void* __restrict__ dstv,
                                                   const float* __restrict__ bias,
                                                   const float* __restrict__ residual) {
  __shared__ __attribute__((aligned(16))) u16b sa[2][128 * 40];
  __shared__ __attribute__((aligned(16))) u16b sb[2][256 * 40];
  const int lane = threadIdx.x & 31;
  const int wave = threadIdx.x >> 5;
  const int wm = wave >> 2;                 // 0..1
  const int wn = wave & 3;                  // 0..3
  const int m0 = blockIdx.y * 128;
  const int n0 = blockIdx.x * 256;

  v8f acc[4][4];
#pragma unroll
  for (int i = 0; i < 4; ++i)
#pragma unroll
    for (int j = 0; j < 4; ++j) acc[i][j] = zero8();

#if USE_TDM
  if (wave == 0) {
    tdm_load_2d(lds_addr_of(&sa[0][0]), (const char*)A + (size_t)m0 * K * 2, K, 128);
    tdm_load_2d(lds_addr_of(&sb[0][0]), (const char*)Bt + (size_t)n0 * K * 2, K, 256);
  }
  int cur = 0;
  for (int kk = 0; kk < K; kk += 32, cur ^= 1) {
    if (wave == 0) __builtin_amdgcn_s_wait_tensorcnt(0);  // current tiles landed
    __syncthreads();
    if (wave == 0 && kk + 32 < K) {                       // prefetch next tiles
      tdm_load_2d(lds_addr_of(&sa[cur ^ 1][0]),
                  (const char*)A + ((size_t)m0 * K + kk + 32) * 2, K, 128);
      tdm_load_2d(lds_addr_of(&sb[cur ^ 1][0]),
                  (const char*)Bt + ((size_t)n0 * K + kk + 32) * 2, K, 256);
    }
    v16bf af[4], bfr[4];
#pragma unroll
    for (int im = 0; im < 4; ++im)
      af[im] = load_a_frag(&sa[cur][(wm * 64 + im * 16) * 40], 40, lane);
#pragma unroll
    for (int jn = 0; jn < 4; ++jn)
      bfr[jn] = load_b_frag(&sb[cur][(wn * 64 + jn * 16) * 40], 40, lane);
#pragma unroll
    for (int im = 0; im < 4; ++im)
#pragma unroll
      for (int jn = 0; jn < 4; ++jn)
        acc[im][jn] = wmma_bf16(af[im], bfr[jn], acc[im][jn]);
  }
#else
  for (int kk = 0; kk < K; kk += 32) {
    __syncthreads();
#pragma unroll
    for (int it = 0; it < 2; ++it) {        // A tile 128x32
      int c = threadIdx.x + it * 256;
      int row = c >> 2, wi = c & 3;
      *(uint4*)&sa[0][row * 40 + wi * 8] =
          *(const uint4*)&A[(size_t)(m0 + row) * K + kk + wi * 8];
    }
#pragma unroll
    for (int it = 0; it < 4; ++it) {        // Bt tile 256x32
      int c = threadIdx.x + it * 256;
      int row = c >> 2, wi = c & 3;
      *(uint4*)&sb[0][row * 40 + wi * 8] =
          *(const uint4*)&Bt[(size_t)(n0 + row) * K + kk + wi * 8];
    }
    __syncthreads();
    v16bf af[4], bfr[4];
#pragma unroll
    for (int im = 0; im < 4; ++im)
      af[im] = load_a_frag(&sa[0][(wm * 64 + im * 16) * 40], 40, lane);
#pragma unroll
    for (int jn = 0; jn < 4; ++jn)
      bfr[jn] = load_b_frag(&sb[0][(wn * 64 + jn * 16) * 40], 40, lane);
#pragma unroll
    for (int im = 0; im < 4; ++im)
#pragma unroll
      for (int jn = 0; jn < 4; ++jn)
        acc[im][jn] = wmma_bf16(af[im], bfr[jn], acc[im][jn]);
  }
#endif

#pragma unroll
  for (int im = 0; im < 4; ++im) {
#pragma unroll
    for (int jn = 0; jn < 4; ++jn) {
#pragma unroll
      for (int r = 0; r < 8; ++r) {
        int row = m0 + wm * 64 + im * 16 + r + ((lane & 16) ? 8 : 0);
        int col = n0 + wn * 64 + jn * 16 + (lane & 15);
        float vacc = acc[im][jn][r];
        if (EPI == 0 || EPI == 1) {          // Q/K -> [b,h,n,hd]
          if (EPI == 0) vacc *= 0.125f;      // 1/sqrt(64)
          int bb = row >> 12, seq = row & 4095;
          int hh = col >> 6,  hd  = col & 63;
          ((u16b*)dstv)[(((size_t)bb * HEADS + hh) * SEQ + seq) * HDIM + hd] =
              f32_to_bf16(vacc);
        } else if (EPI == 2) {               // V -> [b,h,hd,n] (transposed)
          int bb = row >> 12, seq = row & 4095;
          int hh = col >> 6,  hd  = col & 63;
          ((u16b*)dstv)[(((size_t)bb * HEADS + hh) * HDIM + hd) * SEQ + seq] =
              f32_to_bf16(vacc);
        } else if (EPI == 3) {               // residual + acc + bias (f32)
          ((float*)dstv)[(size_t)row * N + col] =
              residual[(size_t)row * N + col] + vacc + bias[col];
        } else {                             // exact GELU (bf16)
          float t = vacc + bias[col];
          float ge = 0.5f * t * (1.0f + erff(t * 0.70710678118654752f));
          ((u16b*)dstv)[(size_t)row * N + col] = f32_to_bf16(ge);
        }
      }
    }
  }
}

// ---------------- Flash attention: 4 waves/block, wave = 32 q-rows, stream 64-key blocks
__global__ __launch_bounds__(128) void attn_kernel(const u16b* __restrict__ q,
                                                   const u16b* __restrict__ k,
                                                   const u16b* __restrict__ v,
                                                   u16b* __restrict__ ao) {
  __shared__ __attribute__((aligned(16))) u16b lds_p[4][32 * 72];
  const int lane  = threadIdx.x & 31;
  const int wave  = threadIdx.x >> 5;
  const int qt    = blockIdx.x & 31;        // 32 q-tiles of 128 per (b,h)
  const int bh    = blockIdx.x >> 5;        // 0..15
  const int qrow0 = qt * 128 + wave * 32;
  const u16b* qbh = q + (size_t)bh * SEQ * HDIM;
  const u16b* kbh = k + (size_t)bh * SEQ * HDIM;
  const u16b* vbh = v + (size_t)bh * HDIM * SEQ;   // [hd][n]
  u16b* pbuf = lds_p[wave];

  v16bf qa[2][2];
#pragma unroll
  for (int i = 0; i < 2; ++i)
#pragma unroll
    for (int kc = 0; kc < 2; ++kc)
      qa[i][kc] = load_a_frag(qbh + (size_t)(qrow0 + i * 16) * HDIM + kc * 32,
                              HDIM, lane);

  v8f o[2][4];
  float m[2][8], l[2][8];
#pragma unroll
  for (int i = 0; i < 2; ++i) {
#pragma unroll
    for (int jh = 0; jh < 4; ++jh) o[i][jh] = zero8();
#pragma unroll
    for (int r = 0; r < 8; ++r) { m[i][r] = -1e30f; l[i][r] = 0.0f; }
  }

  for (int kb = 0; kb < SEQ / 64; ++kb) {
    // ---- S = Q @ K^T for a 64-key block
    v16bf kf[4][2];
#pragma unroll
    for (int jn = 0; jn < 4; ++jn)
#pragma unroll
      for (int kc = 0; kc < 2; ++kc)
        kf[jn][kc] = load_b_frag(kbh + (size_t)(kb * 64 + jn * 16) * HDIM + kc * 32,
                                 HDIM, lane);
    v8f s[2][4];
#pragma unroll
    for (int i = 0; i < 2; ++i)
#pragma unroll
      for (int jn = 0; jn < 4; ++jn) {
        v8f a0 = zero8();
        a0 = wmma_bf16(qa[i][0], kf[jn][0], a0);
        a0 = wmma_bf16(qa[i][1], kf[jn][1], a0);
        s[i][jn] = a0;
      }

    // ---- streaming softmax (row reductions across 16-lane halves)
#pragma unroll
    for (int i = 0; i < 2; ++i) {
#pragma unroll
      for (int r = 0; r < 8; ++r) {
        float mx = s[i][0][r];
#pragma unroll
        for (int jn = 1; jn < 4; ++jn) mx = fmaxf(mx, s[i][jn][r]);
#pragma unroll
        for (int off = 1; off < 16; off <<= 1) mx = fmaxf(mx, __shfl_xor(mx, off));
        float mn = fmaxf(m[i][r], mx);
        float sc = __expf(m[i][r] - mn);
        m[i][r] = mn;
        float sum = 0.0f;
#pragma unroll
        for (int jn = 0; jn < 4; ++jn) {
          float p = __expf(s[i][jn][r] - mn);
          s[i][jn][r] = p;
          sum += p;
        }
#pragma unroll
        for (int off = 1; off < 16; off <<= 1) sum += __shfl_xor(sum, off);
        l[i][r] = l[i][r] * sc + sum;
#pragma unroll
        for (int jh = 0; jh < 4; ++jh) o[i][jh][r] *= sc;
        int prow = i * 16 + r + ((lane & 16) ? 8 : 0);
#pragma unroll
        for (int jn = 0; jn < 4; ++jn)
          pbuf[prow * 72 + jn * 16 + (lane & 15)] = f32_to_bf16(s[i][jn][r]);
      }
    }
    asm volatile("s_wait_dscnt 0" ::: "memory");   // P stores visible before reload

    // ---- O += P @ V
#pragma unroll
    for (int kc2 = 0; kc2 < 2; ++kc2) {
      v16bf pa[2];
#pragma unroll
      for (int i = 0; i < 2; ++i)
        pa[i] = load_a_frag(pbuf + (i * 16) * 72 + kc2 * 32, 72, lane);
#pragma unroll
      for (int jh = 0; jh < 4; ++jh) {
        v16bf vbfr = load_b_frag(vbh + (size_t)(jh * 16) * SEQ + kb * 64 + kc2 * 32,
                                 SEQ, lane);
#pragma unroll
        for (int i = 0; i < 2; ++i) o[i][jh] = wmma_bf16(pa[i], vbfr, o[i][jh]);
      }
    }
  }

  // ---- finalize: divide by l, write [t, h*64+hd] bf16
  const int bb = bh >> 3, hh = bh & 7;
#pragma unroll
  for (int i = 0; i < 2; ++i) {
#pragma unroll
    for (int r = 0; r < 8; ++r) {
      float inv = 1.0f / l[i][r];
      int row = qrow0 + i * 16 + r + ((lane & 16) ? 8 : 0);
      size_t t = (size_t)bb * SEQ + row;
#pragma unroll
      for (int jh = 0; jh < 4; ++jh) {
        int col = hh * 64 + jh * 16 + (lane & 15);
        ao[t * DIMD + col] = f32_to_bf16(o[i][jh][r] * inv);
      }
    }
  }
}

// ---------------- host orchestration
extern "C" void kernel_launch(void* const* d_in, const int* in_sizes, int n_in,
                              void* d_out, int out_size, void* d_ws, size_t ws_size,
                              hipStream_t stream) {
  (void)in_sizes; (void)n_in; (void)out_size; (void)ws_size;
  const float* x   = (const float*)d_in[0];
  const float* Wq  = (const float*)d_in[1];
  const float* Wk  = (const float*)d_in[2];
  const float* Wv  = (const float*)d_in[3];
  const float* Wo  = (const float*)d_in[4];
  const float* bo  = (const float*)d_in[5];
  const float* W1  = (const float*)d_in[6];
  const float* b1  = (const float*)d_in[7];
  const float* W2  = (const float*)d_in[8];
  const float* b2  = (const float*)d_in[9];
  const float* g1  = (const float*)d_in[10];
  const float* be1 = (const float*)d_in[11];
  const float* g2  = (const float*)d_in[12];
  const float* be2 = (const float*)d_in[13];

  char* wsp = (char*)d_ws;
  size_t off = 0;
  auto alloc = [&](size_t bytes) {
    void* p = wsp + off;
    off += (bytes + 255) & ~(size_t)255;
    return p;
  };
  u16b* WqT = (u16b*)alloc((size_t)DIMD * DIMD * 2);
  u16b* WkT = (u16b*)alloc((size_t)DIMD * DIMD * 2);
  u16b* WvT = (u16b*)alloc((size_t)DIMD * DIMD * 2);
  u16b* WoT = (u16b*)alloc((size_t)DIMD * DIMD * 2);
  u16b* W1T = (u16b*)alloc((size_t)DIMD * MLPD * 2);
  u16b* W2T = (u16b*)alloc((size_t)MLPD * DIMD * 2);
  u16b* hb  = (u16b*)alloc((size_t)TOK * DIMD * 2);
  u16b* qb  = (u16b*)alloc((size_t)TOK * DIMD * 2);
  u16b* kb  = (u16b*)alloc((size_t)TOK * DIMD * 2);
  u16b* vb  = (u16b*)alloc((size_t)TOK * DIMD * 2);
  u16b* aob = (u16b*)alloc((size_t)TOK * DIMD * 2);
  float* x1 = (float*)alloc((size_t)TOK * DIMD * 4);
  u16b* h2  = (u16b*)alloc((size_t)TOK * DIMD * 2);
  u16b* ff1 = (u16b*)alloc((size_t)TOK * MLPD * 2);

  // 1) weights -> bf16, transposed to [N][K]
  cvt_t_kernel<<<(DIMD * DIMD) / 256, 256, 0, stream>>>(Wq, WqT, DIMD, DIMD);
  cvt_t_kernel<<<(DIMD * DIMD) / 256, 256, 0, stream>>>(Wk, WkT, DIMD, DIMD);
  cvt_t_kernel<<<(DIMD * DIMD) / 256, 256, 0, stream>>>(Wv, WvT, DIMD, DIMD);
  cvt_t_kernel<<<(DIMD * DIMD) / 256, 256, 0, stream>>>(Wo, WoT, DIMD, DIMD);
  cvt_t_kernel<<<(DIMD * MLPD) / 256, 256, 0, stream>>>(W1, W1T, DIMD, MLPD);
  cvt_t_kernel<<<(MLPD * DIMD) / 256, 256, 0, stream>>>(W2, W2T, MLPD, DIMD);

  // 2) LN1
  ln_kernel<<<TOK / 8, 256, 0, stream>>>(x, g1, be1, hb);

  // 3) QKV projections with scatter epilogues
  dim3 g512(DIMD / 256, TOK / 128);
  gemm_kernel<0><<<g512, 256, 0, stream>>>(hb, WqT, DIMD, DIMD, qb, nullptr, nullptr);
  gemm_kernel<1><<<g512, 256, 0, stream>>>(hb, WkT, DIMD, DIMD, kb, nullptr, nullptr);
  gemm_kernel<2><<<g512, 256, 0, stream>>>(hb, WvT, DIMD, DIMD, vb, nullptr, nullptr);

  // 4) flash attention
  attn_kernel<<<BATCH * HEADS * (SEQ / 128), 128, 0, stream>>>(qb, kb, vb, aob);

  // 5) output projection + residual (f32)
  gemm_kernel<3><<<g512, 256, 0, stream>>>(aob, WoT, DIMD, DIMD, x1, bo, x);

  // 6) LN2
  ln_kernel<<<TOK / 8, 256, 0, stream>>>(x1, g2, be2, h2);

  // 7) MLP up + GELU
  dim3 g2048(MLPD / 256, TOK / 128);
  gemm_kernel<4><<<g2048, 256, 0, stream>>>(h2, W1T, DIMD, MLPD, ff1, b1, nullptr);

  // 8) MLP down + residual -> d_out (f32)
  gemm_kernel<3><<<g512, 256, 0, stream>>>(ff1, W2T, MLPD, DIMD, d_out, b2, x1);
}
